// NBEATS_LSTM_block_10814727652041
// MI455X (gfx1250) — compile-verified, matrix operand order checked by reading, not run on hardware
//
#include <hip/hip_runtime.h>

typedef __attribute__((ext_vector_type(16))) __bf16 v16bf;
typedef __attribute__((ext_vector_type(8)))  __bf16 v8bf;
typedef __attribute__((ext_vector_type(8)))  float  v8f;

#define B_SZ     256
#define L_SZ     512
#define IN_D     64
#define UNITS    256
#define GATES    1024
#define OUTSTEPS 64
#define OUT_D    32
#define KW       5

// ---------------------------------------------------------------------------
// WMMA fragment helpers (wave32, 16x16x32 bf16, f32 accum).
// 16-bit fragment layout (ISA 7.12.2): lane holds elements k0+kb+[0..7] and
// k0+kb+16+[0..7] (kb = (lane>=16)*8). Both runs are contiguous -> a fragment
// is exactly two 16-byte loads when the source is k-contiguous. All B (weight)
// matrices are pre-transposed to [N][K] so this holds for A and B alike.
// C 16x16 f32: lane l elem r -> m = ((l>=16)?8:0)+r, n = l&15.
// ---------------------------------------------------------------------------
__device__ __forceinline__ v16bf frag_contig(const __bf16* p) {
  v8bf lo = *(const v8bf*)p;
  v8bf hi = *(const v8bf*)(p + 16);
  return __builtin_shufflevector(lo, hi, 0, 1, 2, 3, 4, 5, 6, 7,
                                 8, 9, 10, 11, 12, 13, 14, 15);
}

__device__ __forceinline__ v8bf zerov8() {
  v8bf z;
#pragma unroll
  for (int j = 0; j < 8; ++j) z[j] = (__bf16)0.f;
  return z;
}

__device__ __forceinline__ v8f zero8() {
  v8f z = {0.f, 0.f, 0.f, 0.f, 0.f, 0.f, 0.f, 0.f};
  return z;
}

__device__ __forceinline__ float sigmoid_f(float x) {
  return 1.f / (1.f + __expf(-x));
}

#define WMMA_BF16(a, b, c) \
  __builtin_amdgcn_wmma_f32_16x16x32_bf16(false, (a), false, (b), (short)0, (c), false, false)

// ---------------------------------------------------------------------------
// CDNA5 async global->LDS copy (VGLOBAL GLOBAL_LOAD_ASYNC_TO_LDS_B128,
// tracked by ASYNCcnt). LDS byte address = low 32 bits of the generic pointer
// (ISA: LDS aperture truncates addr[31:0]).
// ---------------------------------------------------------------------------
__device__ __forceinline__ void async_copy16(__bf16* lds_dst, const __bf16* gsrc) {
  unsigned dst = (unsigned)(size_t)lds_dst;
  unsigned long long src = (unsigned long long)(size_t)gsrc;
  asm volatile("global_load_async_to_lds_b128 %0, %1, off"
               :: "v"(dst), "v"(src) : "memory");
}

__device__ __forceinline__ void async_wait0() {
  asm volatile("s_wait_asynccnt 0x0" ::: "memory");
}

// ---------------------------------------------------------------------------
// Prep kernels: f32 -> bf16 (optionally transposed to [N][K], or summed)
// ---------------------------------------------------------------------------
__global__ void __launch_bounds__(256) cvt_kernel(const float* __restrict__ s,
                                                  __bf16* __restrict__ d, int n) {
  int i = blockIdx.x * 256 + threadIdx.x;
  if (i < n) d[i] = (__bf16)s[i];
}

// s: [K][N] row-major  ->  d: [N][K]
__global__ void __launch_bounds__(256) cvtT_kernel(const float* __restrict__ s,
                                                   __bf16* __restrict__ d, int K, int N) {
  int i = blockIdx.x * 256 + threadIdx.x;
  if (i < K * N) {
    int k = i / N, n = i - k * N;
    d[(size_t)n * K + k] = (__bf16)s[i];
  }
}

// d[N][K] = (a + b)^T
__global__ void __launch_bounds__(256) addcvtT_kernel(const float* __restrict__ a,
                                                      const float* __restrict__ b,
                                                      __bf16* __restrict__ d, int K, int N) {
  int i = blockIdx.x * 256 + threadIdx.x;
  if (i < K * N) {
    int k = i / N, n = i - k * N;
    d[(size_t)n * K + k] = (__bf16)(a[i] + b[i]);
  }
}

// ---------------------------------------------------------------------------
// Conv1D 'same' K=5 as 10 accumulated WMMA k-chunks, + bias + relu.
// CKt: [256][320] (unit-major, k = kk*64+i). 16 rows x 128 cols per block.
// ---------------------------------------------------------------------------
__global__ void __launch_bounds__(256) conv_gemm_kernel(const __bf16* __restrict__ Xin,
                                                        const __bf16* __restrict__ CKt,
                                                        const float* __restrict__ bias,
                                                        __bf16* __restrict__ Xout) {
  __shared__ alignas(16) __bf16 As[16][40];
  __shared__ alignas(16) __bf16 Bs[128][40];
  const int tid = threadIdx.x, wave = tid >> 5, lane = tid & 31;
  const int row0 = blockIdx.x * 16;
  const int col0 = blockIdx.y * 128;
  const int b = row0 >> 9, l0 = row0 & (L_SZ - 1);
  const int kb = (lane >> 4) * 8;
  v8f acc = zero8();
  for (int kk = 0; kk < KW; ++kk) {
    for (int kc = 0; kc < 2; ++kc) {
      // A: 16 rows x 32 k = 64 chunks of 8 bf16, row-level bounds check
      for (int i = tid; i < 64; i += 256) {
        int m = i >> 2, ch = (i & 3) * 8;
        int l = l0 + m + kk - 2;
        v8bf val = zerov8();
        if (l >= 0 && l < L_SZ)
          val = *(const v8bf*)(Xin + ((size_t)b * L_SZ + l) * IN_D + kc * 32 + ch);
        *(v8bf*)&As[m][ch] = val;
      }
      // B: 128 cols x 32 k = 512 chunks
      for (int i = tid; i < 512; i += 256) {
        int c = i >> 2, ch = (i & 3) * 8;
        *(v8bf*)&Bs[c][ch] =
            *(const v8bf*)(CKt + (size_t)(col0 + c) * (KW * IN_D) + kk * IN_D + kc * 32 + ch);
      }
      __syncthreads();
      v16bf a = frag_contig(&As[lane & 15][kb]);
      v16bf bb = frag_contig(&Bs[wave * 16 + (lane & 15)][kb]);
      acc = WMMA_BF16(a, bb, acc);
      __syncthreads();
    }
  }
  const int mb = (lane >> 4) * 8, n = lane & 15;
  const int cc = col0 + wave * 16 + n;
  const float bv = bias[cc];
#pragma unroll
  for (int r = 0; r < 8; ++r) {
    float v = fmaxf(acc[r] + bv, 0.f);
    Xout[(size_t)(row0 + mb + r) * UNITS + cc] = (__bf16)v;
  }
}

// ---------------------------------------------------------------------------
// bf16 WMMA GEMM specialized to K=256: out = A[Mx256] * Wt^T + bias.
// Wt stored [N][256]. Full K panels staged once (async global->LDS), then a
// single barrier and 8 back-to-back WMMAs. Tile 16 x 128, 256 thr = 8 waves.
// ---------------------------------------------------------------------------
template <bool OUT_BF16>
__global__ void __launch_bounds__(256) gemm_k256_kernel(const __bf16* __restrict__ A,
                                                        const __bf16* __restrict__ Wt,
                                                        const float* __restrict__ bias,
                                                        void* __restrict__ out,
                                                        int N) {
  __shared__ alignas(16) __bf16 As[16][264];    // 16 x 256 (+pad)
  __shared__ alignas(16) __bf16 Bs[128][264];   // 128 x 256 (+pad)
  const int tid = threadIdx.x, wave = tid >> 5, lane = tid & 31;
  const int row0 = blockIdx.x * 16;
  const int col0 = blockIdx.y * 128;
  const int kb = (lane >> 4) * 8;

  // A panel: 16 rows x 32 chunks = 512 async 16B copies (always in range)
  for (int i = tid; i < 512; i += 256) {
    int m = i >> 5, ch = (i & 31) * 8;
    async_copy16(&As[m][ch], A + (size_t)(row0 + m) * UNITS + ch);
  }
  // B panel: 128 cols x 32 chunks
  if (col0 + 128 <= N) {
    for (int i = tid; i < 4096; i += 256) {
      int c = i >> 5, ch = (i & 31) * 8;
      async_copy16(&Bs[c][ch], Wt + (size_t)(col0 + c) * UNITS + ch);
    }
  } else {
    // partial tile: synchronous staging with zero fill
    for (int i = tid; i < 4096; i += 256) {
      int c = i >> 5, ch = (i & 31) * 8;
      v8bf val = zerov8();
      if (col0 + c < N) val = *(const v8bf*)(Wt + (size_t)(col0 + c) * UNITS + ch);
      *(v8bf*)&Bs[c][ch] = val;
    }
  }
  async_wait0();
  __syncthreads();

  v8f acc = zero8();
#pragma unroll
  for (int kc = 0; kc < 8; ++kc) {
    v16bf a = frag_contig(&As[lane & 15][kc * 32 + kb]);
    v16bf bb = frag_contig(&Bs[wave * 16 + (lane & 15)][kc * 32 + kb]);
    acc = WMMA_BF16(a, bb, acc);
  }

  const int mb = (lane >> 4) * 8, n = lane & 15;
  const int cc = col0 + wave * 16 + n;
  if (cc < N) {
    const float bv = bias[cc];
#pragma unroll
    for (int r = 0; r < 8; ++r) {
      float v = acc[r] + bv;
      size_t o = (size_t)(row0 + mb + r) * N + cc;
      if (OUT_BF16) ((__bf16*)out)[o] = (__bf16)v;
      else          ((float*)out)[o] = v;
    }
  }
}

// ---------------------------------------------------------------------------
// Encoder LSTM recurrence. 16 blocks x 512 threads (16 waves). Block owns 16
// batch rows for all 512 steps (no inter-WG sync). Wave w owns units u0=w*16
// + all 4 gate tiles for them, so i/f/g/o land in the same lane and c stays in
// registers. Ut: [1024 gate cols][256 k] bf16. Weight tiers: kc0-3 VGPRs,
// kc4-5 LDS (staged once), kc6-7 gathered from L2. h double-buffered in LDS.
// ---------------------------------------------------------------------------
#define HSTR (UNITS + 8)

__global__ void __launch_bounds__(512) enc_lstm_kernel(const __bf16* __restrict__ Zx,
                                                       const __bf16* __restrict__ Ut,
                                                       __bf16* __restrict__ Hseq,
                                                       float* __restrict__ hT,
                                                       float* __restrict__ cT) {
  __shared__ alignas(16) __bf16 h_tile[2][16][HSTR];  // ~17 KB
  __shared__ alignas(16) __bf16 Us[GATES][64];        // 128 KB: k rows 128..191
  const int tid = threadIdx.x, wave = tid >> 5, lane = tid & 31;
  const int b0 = blockIdx.x * 16;
  const int u0 = wave * 16;
  const int n = lane & 15;
  const int mb = (lane >> 4) * 8;
  const int kb = (lane >> 4) * 8;
  const int u = u0 + n;

  // register-persistent U fragments for kc = 0..3 (k rows 0..127)
  v16bf ufrag[4][4];
#pragma unroll
  for (int g = 0; g < 4; ++g)
#pragma unroll
    for (int kc = 0; kc < 4; ++kc)
      ufrag[g][kc] = frag_contig(Ut + (size_t)(g * UNITS + u) * UNITS + kc * 32 + kb);

  // stage U k-rows 128..191 into LDS, [col][k-128] layout (reused all steps)
  for (int i = tid; i < GATES * 8; i += 512) {
    int c = i >> 3, ch = (i & 7) * 8;
    *(v8bf*)&Us[c][ch] = *(const v8bf*)(Ut + (size_t)c * UNITS + 128 + ch);
  }

  float creg[8];
#pragma unroll
  for (int r = 0; r < 8; ++r) creg[r] = 0.f;
  for (int i = tid; i < 16 * HSTR; i += 512) {
    h_tile[0][0][i] = (__bf16)0.f;
    h_tile[1][0][i] = (__bf16)0.f;
  }
  __syncthreads();

  for (int t = 0; t < L_SZ; ++t) {
    const int cur = t & 1, nxt = cur ^ 1;
    v8f acc[4];
#pragma unroll
    for (int g = 0; g < 4; ++g) acc[g] = zero8();

#pragma unroll
    for (int kc = 0; kc < 8; ++kc) {
      v16bf a = frag_contig(&h_tile[cur][lane & 15][kc * 32 + kb]);
#pragma unroll
      for (int g = 0; g < 4; ++g) {
        const int col = g * UNITS + u;
        v16bf bfr;
        if (kc < 4)      bfr = ufrag[g][kc];                                   // VGPR tier
        else if (kc < 6) bfr = frag_contig(&Us[col][(kc - 4) * 32 + kb]);      // LDS tier
        else             bfr = frag_contig(Ut + (size_t)col * UNITS + kc * 32 + kb); // L2 tier
        acc[g] = WMMA_BF16(a, bfr, acc[g]);
      }
    }

#pragma unroll
    for (int r = 0; r < 8; ++r) {
      const size_t row = (size_t)(b0 + mb + r) * L_SZ + t;
      const __bf16* zx = Zx + row * GATES;  // enc_b already folded in
      float zi = acc[0][r] + (float)zx[u];
      float zf = acc[1][r] + (float)zx[u + UNITS];
      float zg = acc[2][r] + (float)zx[u + 2 * UNITS];
      float zo = acc[3][r] + (float)zx[u + 3 * UNITS];
      float cn = sigmoid_f(zf) * creg[r] + sigmoid_f(zi) * tanhf(zg);
      float hn = sigmoid_f(zo) * tanhf(cn);
      creg[r] = cn;
      __bf16 hb = (__bf16)hn;
      h_tile[nxt][mb + r][u] = hb;
      Hseq[row * UNITS + u] = hb;
      if (t + 1 < L_SZ) __builtin_prefetch((const void*)(zx + GATES), 0, 1);
    }
    __syncthreads();
  }

#pragma unroll
  for (int r = 0; r < 8; ++r) {
    const int bi = b0 + mb + r;
    hT[(size_t)bi * UNITS + u] = (float)h_tile[L_SZ & 1][mb + r][u];
    cT[(size_t)bi * UNITS + u] = creg[r];
  }
}

// ---------------------------------------------------------------------------
// Decoder: x_t == h  =>  z = h @ (cell_W + cell_U) + cell_b. Same structure.
// WcT: [1024][256] bf16 (pre-summed + transposed).
// ---------------------------------------------------------------------------
__global__ void __launch_bounds__(512) dec_lstm_kernel(const __bf16* __restrict__ WcT,
                                                       const float* __restrict__ bias,
                                                       const float* __restrict__ h0,
                                                       const float* __restrict__ c0,
                                                       __bf16* __restrict__ Dseq) {
  __shared__ alignas(16) __bf16 h_tile[2][16][HSTR];
  __shared__ alignas(16) __bf16 Us[GATES][64];
  const int tid = threadIdx.x, wave = tid >> 5, lane = tid & 31;
  const int b0 = blockIdx.x * 16;
  const int u0 = wave * 16;
  const int n = lane & 15;
  const int mb = (lane >> 4) * 8;
  const int kb = (lane >> 4) * 8;
  const int u = u0 + n;

  v16bf ufrag[4][4];
#pragma unroll
  for (int g = 0; g < 4; ++g)
#pragma unroll
    for (int kc = 0; kc < 4; ++kc)
      ufrag[g][kc] = frag_contig(WcT + (size_t)(g * UNITS + u) * UNITS + kc * 32 + kb);

  for (int i = tid; i < GATES * 8; i += 512) {
    int c = i >> 3, ch = (i & 7) * 8;
    *(v8bf*)&Us[c][ch] = *(const v8bf*)(WcT + (size_t)c * UNITS + 128 + ch);
  }

  const float bi_ = bias[u];
  const float bf_ = bias[u + UNITS];
  const float bg_ = bias[u + 2 * UNITS];
  const float bo_ = bias[u + 3 * UNITS];

  float creg[8];
#pragma unroll
  for (int r = 0; r < 8; ++r)
    creg[r] = c0[(size_t)(b0 + mb + r) * UNITS + u];
  for (int i = tid; i < 16 * UNITS; i += 512) {
    int m = i >> 8, uu = i & 255;
    h_tile[0][m][uu] = (__bf16)h0[(size_t)(b0 + m) * UNITS + uu];
  }
  __syncthreads();

  for (int t = 0; t < OUTSTEPS; ++t) {
    const int cur = t & 1, nxt = cur ^ 1;
    v8f acc[4];
#pragma unroll
    for (int g = 0; g < 4; ++g) acc[g] = zero8();

#pragma unroll
    for (int kc = 0; kc < 8; ++kc) {
      v16bf a = frag_contig(&h_tile[cur][lane & 15][kc * 32 + kb]);
#pragma unroll
      for (int g = 0; g < 4; ++g) {
        const int col = g * UNITS + u;
        v16bf bfr;
        if (kc < 4)      bfr = ufrag[g][kc];
        else if (kc < 6) bfr = frag_contig(&Us[col][(kc - 4) * 32 + kb]);
        else             bfr = frag_contig(WcT + (size_t)col * UNITS + kc * 32 + kb);
        acc[g] = WMMA_BF16(a, bfr, acc[g]);
      }
    }

#pragma unroll
    for (int r = 0; r < 8; ++r) {
      float zi = acc[0][r] + bi_;
      float zf = acc[1][r] + bf_;
      float zg = acc[2][r] + bg_;
      float zo = acc[3][r] + bo_;
      float cn = sigmoid_f(zf) * creg[r] + sigmoid_f(zi) * tanhf(zg);
      float hn = sigmoid_f(zo) * tanhf(cn);
      creg[r] = cn;
      __bf16 hb = (__bf16)hn;
      h_tile[nxt][mb + r][u] = hb;
      Dseq[((size_t)(b0 + mb + r) * OUTSTEPS + t) * UNITS + u] = hb;
    }
    __syncthreads();
  }
}

// ---------------------------------------------------------------------------
// Host-side orchestration
// ---------------------------------------------------------------------------
extern "C" void kernel_launch(void* const* d_in, const int* in_sizes, int n_in,
                              void* d_out, int out_size, void* d_ws, size_t ws_size,
                              hipStream_t stream) {
  (void)in_sizes; (void)n_in; (void)out_size; (void)ws_size;
  const float* inputs = (const float*)d_in[0];
  const float* conv_k = (const float*)d_in[1];
  const float* conv_b = (const float*)d_in[2];
  const float* enc_W  = (const float*)d_in[3];
  const float* enc_U  = (const float*)d_in[4];
  const float* enc_b  = (const float*)d_in[5];
  const float* cell_W = (const float*)d_in[6];
  const float* cell_U = (const float*)d_in[7];
  const float* cell_b = (const float*)d_in[8];
  const float* fore_W = (const float*)d_in[9];
  const float* fore_b = (const float*)d_in[10];
  const float* back_W = (const float*)d_in[11];
  const float* back_b = (const float*)d_in[12];

  char* ws = (char*)d_ws;
  size_t off = 0;
  auto wsalloc = [&](size_t bytes) -> void* {
    void* p = ws + off;
    off += (bytes + 255) & ~(size_t)255;
    return p;
  };
  __bf16* Xin_bf = (__bf16*)wsalloc((size_t)B_SZ * L_SZ * IN_D * 2);
  __bf16* Xbf    = (__bf16*)wsalloc((size_t)B_SZ * L_SZ * UNITS * 2);
  __bf16* Zx     = (__bf16*)wsalloc((size_t)B_SZ * L_SZ * GATES * 2);
  __bf16* Hseq   = (__bf16*)wsalloc((size_t)B_SZ * L_SZ * UNITS * 2);
  __bf16* Dseq   = (__bf16*)wsalloc((size_t)B_SZ * OUTSTEPS * UNITS * 2);
  float*  hT     = (float*)wsalloc((size_t)B_SZ * UNITS * 4);
  float*  cT     = (float*)wsalloc((size_t)B_SZ * UNITS * 4);
  __bf16* encWt  = (__bf16*)wsalloc((size_t)GATES * UNITS * 2);     // [1024][256]
  __bf16* encUt  = (__bf16*)wsalloc((size_t)GATES * UNITS * 2);     // [1024][256]
  __bf16* cellWt = (__bf16*)wsalloc((size_t)GATES * UNITS * 2);     // [1024][256]
  __bf16* CKt    = (__bf16*)wsalloc((size_t)UNITS * KW * IN_D * 2); // [256][320]
  __bf16* foreWt = (__bf16*)wsalloc((size_t)OUT_D * UNITS * 2);     // [32][256]
  __bf16* backWt = (__bf16*)wsalloc((size_t)IN_D * UNITS * 2);      // [64][256]

  // Weight prep (transpose to [N][K] bf16)
  cvt_kernel<<<(B_SZ * L_SZ * IN_D + 255) / 256, 256, 0, stream>>>(
      inputs, Xin_bf, B_SZ * L_SZ * IN_D);
  cvtT_kernel<<<(KW * IN_D * UNITS + 255) / 256, 256, 0, stream>>>(
      conv_k, CKt, KW * IN_D, UNITS);
  cvtT_kernel<<<(UNITS * GATES + 255) / 256, 256, 0, stream>>>(enc_W, encWt, UNITS, GATES);
  cvtT_kernel<<<(UNITS * GATES + 255) / 256, 256, 0, stream>>>(enc_U, encUt, UNITS, GATES);
  cvtT_kernel<<<(UNITS * OUT_D + 255) / 256, 256, 0, stream>>>(fore_W, foreWt, UNITS, OUT_D);
  cvtT_kernel<<<(UNITS * IN_D + 255) / 256, 256, 0, stream>>>(back_W, backWt, UNITS, IN_D);
  addcvtT_kernel<<<(UNITS * GATES + 255) / 256, 256, 0, stream>>>(
      cell_W, cell_U, cellWt, UNITS, GATES);

  // Conv1D + relu -> Xbf  [BL x 256]
  conv_gemm_kernel<<<dim3(B_SZ * L_SZ / 16, 2), 256, 0, stream>>>(
      Xin_bf, CKt, conv_b, Xbf);

  // Zx = Xbf @ enc_W + enc_b  (bf16), whole-device GEMM (async-staged panels)
  gemm_k256_kernel<true><<<dim3(B_SZ * L_SZ / 16, GATES / 128), 256, 0, stream>>>(
      Xbf, encWt, enc_b, (void*)Zx, GATES);

  // Recurrences (16 persistent WGs each; batch-parallel, no inter-WG sync)
  enc_lstm_kernel<<<B_SZ / 16, 512, 0, stream>>>(Zx, encUt, Hseq, hT, cT);
  dec_lstm_kernel<<<B_SZ / 16, 512, 0, stream>>>(cellWt, cell_b, hT, cT, Dseq);

  // Output projections (f32 into d_out: forecast then backcast)
  float* fore_out = (float*)d_out;
  float* back_out = (float*)d_out + (size_t)B_SZ * OUTSTEPS * OUT_D;
  gemm_k256_kernel<false><<<dim3(B_SZ * OUTSTEPS / 16, 1), 256, 0, stream>>>(
      Dseq, foreWt, fore_b, (void*)fore_out, OUT_D);
  gemm_k256_kernel<false><<<dim3(B_SZ * L_SZ / 16, 1), 256, 0, stream>>>(
      Hseq, backWt, back_b, (void*)back_out, IN_D);
}
